// MultiScaleDefAttn2D_85289460564230
// MI455X (gfx1250) — matrix operand order repaired and checked by reading, not compiled, not run on hardware
//
#include <hip/hip_runtime.h>
#include <hip/hip_bf16.h>
#include <math.h>

typedef __attribute__((ext_vector_type(16))) _Float16 v16h;
typedef __attribute__((ext_vector_type(8)))  _Float16 v8h;
typedef __attribute__((ext_vector_type(8)))  float    v8f;

#define NQ_     2500
#define E_      256
#define CAM_    6
#define HEADS_  8
#define HDIM_   32
#define LVLS_   4
#define PTS_    4
#define GEMM1N  384              // 256 offset cols + 128 weight-logit cols
#define TILES_M 157              // ceil(2500/16)

// ---------------------------------------------------------------------------
// Prep: f16 conversions / transposes so the WMMA fragment loads are contiguous
// ---------------------------------------------------------------------------
__global__ __launch_bounds__(256)
void pack_query_f16(const float* __restrict__ q, _Float16* __restrict__ qh)
{
    int i = blockIdx.x * blockDim.x + threadIdx.x;
    if (i < NQ_ * E_) qh[i] = (_Float16)q[i];
}

// Wcat_t[n][k] = n<256 ? W_offset[k][n] : W_weight[k][n-256]   (f16, transposed)
__global__ __launch_bounds__(256)
void pack_wcat_t(const float* __restrict__ Woff, const float* __restrict__ Wwt,
                 _Float16* __restrict__ Wt)
{
    int i = blockIdx.x * blockDim.x + threadIdx.x;
    if (i >= GEMM1N * E_) return;
    int n = i >> 8;
    int k = i & 255;
    float v = (n < 256) ? Woff[k * 256 + n] : Wwt[k * 128 + (n - 256)];
    Wt[i] = (_Float16)v;
}

// Wproj_t[n][k] = W_proj[k][n]   (f16, transposed)
__global__ __launch_bounds__(256)
void pack_wproj_t(const float* __restrict__ Wp, _Float16* __restrict__ Wt)
{
    int i = blockIdx.x * blockDim.x + threadIdx.x;
    if (i >= E_ * E_) return;
    int n = i >> 8;
    int k = i & 255;
    Wt[i] = (_Float16)Wp[k * 256 + n];
}

// ---------------------------------------------------------------------------
// Wave-level 16x16 WMMA tile GEMM core over K=256 (fully unrolled, 8 WMMAs).
// A: f16 row-major [*, 256]; Bt: f16 transposed [N][256].
// ---------------------------------------------------------------------------
__device__ __forceinline__ v8f wmma_k256(const _Float16* __restrict__ arow,
                                         const _Float16* __restrict__ bcol,
                                         int grp)
{
    v8f acc = {};
#pragma unroll
    for (int k0 = 0; k0 < E_; k0 += 32) {
        // A fragment: lanes 0-15 K=k0+j, lanes 16-31 K=k0+16+j -> contiguous 16
        v16h a, b;
        a = *(const v16h*)(arow + k0 + 16 * grp);
        // B fragment: lanes 0-15 K=k0+j, lanes 16-31 K=k0+16+j -> contiguous 16
        b = *(const v16h*)(bcol + k0 + 16 * grp);
        acc = __builtin_amdgcn_wmma_f32_16x16x32_f16(
                  false, a, false, b, (short)0, acc, false, false);
    }
    return acc;
}

// ---------------------------------------------------------------------------
// GEMM1: out1[q, 0:256] = q@W_offset + b_offset ; out1[q,256:384] = logits
// ---------------------------------------------------------------------------
__global__ __launch_bounds__(256)
void msda_qproj_wmma(const _Float16* __restrict__ queryh,
                     const _Float16* __restrict__ Wcat_t,
                     const float* __restrict__ b_offset,
                     const float* __restrict__ b_weight,
                     float* __restrict__ out1)
{
    const int tilesN   = GEMM1N / 16;            // 24
    const int numTiles = TILES_M * tilesN;       // 3768
    int wave = (blockIdx.x * blockDim.x + threadIdx.x) >> 5;
    int lane = threadIdx.x & 31;
    if (wave >= numTiles) return;

    int tm  = wave / tilesN;
    int tn  = wave % tilesN;
    int l16 = lane & 15;
    int grp = lane >> 4;
    int row = tm * 16 + l16;
    int col = tn * 16 + l16;
    int rowc = row < NQ_ ? row : (NQ_ - 1);      // clamp: extra D rows never stored

    v8f acc = wmma_k256(queryh + (size_t)rowc * E_,
                        Wcat_t + (size_t)col * E_, grp);

    float bias = (col < 256) ? b_offset[col] : b_weight[col - 256];
#pragma unroll
    for (int r = 0; r < 8; ++r) {
        int m = tm * 16 + r + grp * 8;           // D: VGPR r -> M = r + 8*grp
        if (m < NQ_) out1[m * GEMM1N + col] = acc[r] + bias;
    }
}

// ---------------------------------------------------------------------------
// Softmax over the 16 (level,point) logits per (query, head).
// ---------------------------------------------------------------------------
__global__ __launch_bounds__(256)
void msda_softmax(const float* __restrict__ out1, float* __restrict__ wts)
{
    int idx = blockIdx.x * blockDim.x + threadIdx.x;      // over NQ_*HEADS_
    if (idx >= NQ_ * HEADS_) return;
    int q = idx >> 3;
    int h = idx & 7;
    const float* lg = out1 + q * GEMM1N + 256 + h * 16;
    float mx = lg[0];
#pragma unroll
    for (int j = 1; j < 16; ++j) mx = fmaxf(mx, lg[j]);
    float ex[16];
    float sum = 0.f;
#pragma unroll
    for (int j = 0; j < 16; ++j) { ex[j] = __expf(lg[j] - mx); sum += ex[j]; }
    float inv = 1.f / sum;
    float* w = wts + idx * 16;
#pragma unroll
    for (int j = 0; j < 16; ++j) w[j] = ex[j] * inv;
}

// ---------------------------------------------------------------------------
// Deformable sampling: one wave32 per (query, head); lane = channel (HDIM_=32).
// Branchless bilinear (clamp + in-bounds multiplier), masked camera sum,
// writes slots/count directly in f16 (GEMM2's A operand format).
// ---------------------------------------------------------------------------
__global__ __launch_bounds__(256)
void msda_sample(const float* __restrict__ out1,
                 const float* __restrict__ wts,
                 const float* __restrict__ value,
                 const float* __restrict__ point,
                 const unsigned char* __restrict__ valid,
                 _Float16* __restrict__ slotsh)
{
    const int LVL_H[LVLS_] = {64, 32, 16, 8};
    const int LVL_W[LVLS_] = {176, 88, 44, 22};
    const int LVL_S[LVLS_] = {0, 11264, 14080, 14784};   // level starts in VN

    int wave = (blockIdx.x * blockDim.x + threadIdx.x) >> 5;
    int lane = threadIdx.x & 31;
    if (wave >= NQ_ * HEADS_) return;
    int q = wave >> 3;
    int h = wave & 7;

    const float* offrow = out1 + q * GEMM1N;             // first 256 = offsets
    const float* wrow   = wts + (q * HEADS_ + h) * 16;

    float acc = 0.f;
    float cnt = 0.f;
    for (int cam = 0; cam < CAM_; ++cam) {
        float msk = valid[cam * NQ_ + q] ? 1.f : 0.f;    // valid[cam, 0, q]
        cnt += msk;
        float px = point[(cam * NQ_ + q) * 2 + 0];       // point[cam, 0, q, :]
        float py = point[(cam * NQ_ + q) * 2 + 1];
        float camacc = 0.f;
#pragma unroll
        for (int l = 0; l < LVLS_; ++l) {
            const int   Hl = LVL_H[l], Wl = LVL_W[l], st = LVL_S[l];
            const float fW = (float)Wl, fH = (float)Hl;
#pragma unroll
            for (int p = 0; p < PTS_; ++p) {
                int oi = ((h * LVLS_ + l) * PTS_ + p) * 2;
                // (px + ox/W)*W - 0.5  ==  px*W + ox - 0.5
                float gx = px * fW + offrow[oi + 0] - 0.5f;
                float gy = py * fH + offrow[oi + 1] - 0.5f;
                float x0f = floorf(gx), y0f = floorf(gy);
                float wx = gx - x0f,    wy = gy - y0f;

                float s = 0.f;
#pragma unroll
                for (int dy = 0; dy < 2; ++dy) {
#pragma unroll
                    for (int dx = 0; dx < 2; ++dx) {
                        float xf = x0f + (float)dx;
                        float yf = y0f + (float)dy;
                        float cw = (dx ? wx : (1.f - wx)) * (dy ? wy : (1.f - wy));
                        float inb = (xf >= 0.f) && (xf <= fW - 1.f) &&
                                    (yf >= 0.f) && (yf <= fH - 1.f) ? 1.f : 0.f;
                        float xc = fminf(fmaxf(xf, 0.f), fW - 1.f);
                        float yc = fminf(fmaxf(yf, 0.f), fH - 1.f);
                        int xi = (int)xc, yi = (int)yc;
                        int vn = st + yi * Wl + xi;
                        float v = value[((size_t)vn * CAM_ + cam) * E_
                                        + h * HDIM_ + lane];
                        s += cw * inb * v;
                    }
                }
                camacc += wrow[l * PTS_ + p] * s;
            }
        }
        acc += msk * camacc;
    }
    cnt = fmaxf(cnt, 1.f);
    slotsh[q * E_ + h * HDIM_ + lane] = (_Float16)(acc / cnt);
}

// ---------------------------------------------------------------------------
// GEMM2: out[q, :] = slots @ W_proj + b_proj  -> d_out [1, 2500, 256]
// ---------------------------------------------------------------------------
__global__ __launch_bounds__(256)
void msda_proj_wmma(const _Float16* __restrict__ slotsh,
                    const _Float16* __restrict__ Wproj_t,
                    const float* __restrict__ b_proj,
                    float* __restrict__ out)
{
    const int tilesN   = E_ / 16;                // 16
    const int numTiles = TILES_M * tilesN;       // 2512
    int wave = (blockIdx.x * blockDim.x + threadIdx.x) >> 5;
    int lane = threadIdx.x & 31;
    if (wave >= numTiles) return;

    int tm  = wave / tilesN;
    int tn  = wave % tilesN;
    int l16 = lane & 15;
    int grp = lane >> 4;
    int row = tm * 16 + l16;
    int col = tn * 16 + l16;
    int rowc = row < NQ_ ? row : (NQ_ - 1);

    v8f acc = wmma_k256(slotsh + (size_t)rowc * E_,
                        Wproj_t + (size_t)col * E_, grp);

    float bias = b_proj[col];
#pragma unroll
    for (int r = 0; r < 8; ++r) {
        int m = tm * 16 + r + grp * 8;
        if (m < NQ_) out[m * E_ + col] = acc[r] + bias;
    }
}

// ---------------------------------------------------------------------------
extern "C" void kernel_launch(void* const* d_in, const int* in_sizes, int n_in,
                              void* d_out, int out_size, void* d_ws, size_t ws_size,
                              hipStream_t stream)
{
    const float*         query    = (const float*)d_in[0];
    const float*         value    = (const float*)d_in[1];
    const float*         point    = (const float*)d_in[2];
    const unsigned char* valid    = (const unsigned char*)d_in[3];
    /* d_in[4] shape_hw: compile-time constant, unused on device */
    const float*         W_offset = (const float*)d_in[5];
    const float*         b_offset = (const float*)d_in[6];
    const float*         W_weight = (const float*)d_in[7];
    const float*         b_weight = (const float*)d_in[8];
    const float*         W_proj   = (const float*)d_in[9];
    const float*         b_proj   = (const float*)d_in[10];
    float*               out      = (float*)d_out;

    // Workspace layout:
    //   out1   f32 [NQ,384]        wts    f32 [NQ,8,16]
    //   queryh f16 [NQ,256]        slotsh f16 [NQ,256]
    //   Wcat_t f16 [384,256]       Wproj_t f16 [256,256]
    float*    out1    = (float*)d_ws;
    float*    wts     = out1 + (size_t)NQ_ * GEMM1N;
    _Float16* queryh  = (_Float16*)(wts + (size_t)NQ_ * (HEADS_ * 16));
    _Float16* slotsh  = queryh + (size_t)NQ_ * E_;
    _Float16* Wcat_t  = slotsh + (size_t)NQ_ * E_;
    _Float16* Wproj_t = Wcat_t + (size_t)GEMM1N * E_;

    // 0) Pack operands into WMMA-friendly f16 layouts
    pack_query_f16<<<(NQ_ * E_ + 255) / 256, 256, 0, stream>>>(query, queryh);
    pack_wcat_t<<<(GEMM1N * E_ + 255) / 256, 256, 0, stream>>>(W_offset, W_weight, Wcat_t);
    pack_wproj_t<<<(E_ * E_ + 255) / 256, 256, 0, stream>>>(W_proj, Wproj_t);

    // 1) Fused offset+weight projection GEMM (query is camera-broadcast)
    {
        int tiles  = TILES_M * (GEMM1N / 16);            // 3768 waves
        int blocks = (tiles + 7) / 8;                    // 8 waves / block
        msda_qproj_wmma<<<blocks, 256, 0, stream>>>(
            queryh, Wcat_t, b_offset, b_weight, out1);
    }

    // 2) Softmax over 16 logits per (q, head)
    {
        int n = NQ_ * HEADS_;
        msda_softmax<<<(n + 255) / 256, 256, 0, stream>>>(out1, wts);
    }

    // 3) Deformable bilinear sampling + masked camera reduction -> f16 slots
    {
        int waves  = NQ_ * HEADS_;                       // 20000
        int blocks = (waves + 7) / 8;
        msda_sample<<<blocks, 256, 0, stream>>>(
            out1, wts, value, point, valid, slotsh);
    }

    // 4) Output projection GEMM -> d_out
    {
        int tiles  = TILES_M * (E_ / 16);                // 2512 waves
        int blocks = (tiles + 7) / 8;
        msda_proj_wmma<<<blocks, 256, 0, stream>>>(slotsh, Wproj_t, b_proj, out);
    }
}
